// RNN_custom_3393024164608
// MI455X (gfx1250) — matrix operand order repaired and checked by reading, not compiled, not run on hardware
//
#include <hip/hip_runtime.h>
#include <hip/hip_bf16.h>

// ---------------- problem sizes ----------------
#define H_    256          // hidden
#define IN_   256          // input
#define G3    768          // 3*H (gates)
#define BB    128          // batch
#define TT    1024         // timesteps
#define OUTN  128          // output
#define MTOT  (BB*TT)      // 131072 rows of x

// persistent recurrent kernel geometry: 48 blocks * 8 wave32 = 384 waves
// = exactly one 16x16 tile per wave for a (128 x 768) gate matmul.
// Mapping: block b owns n-tile column b; wave w in block owns m-tile w.
#define RNB       48
#define RNT       256
#define RTHREADS  (RNB*RNT)

typedef __attribute__((ext_vector_type(16))) _Float16 v16h;
typedef __attribute__((ext_vector_type(8)))  float    v8f;
typedef __attribute__((ext_vector_type(4)))  unsigned v4u32;
typedef __attribute__((ext_vector_type(8)))  int      v8i32;
typedef __attribute__((ext_vector_type(4)))  int      v4i32;

#if defined(__has_builtin)
#if __has_builtin(__builtin_amdgcn_tensor_load_to_lds)
#define HAVE_TDM 1
#endif
#endif
#ifndef HAVE_TDM
#define HAVE_TDM 0
#endif

// ---- CDNA5 WMMA f16 fragment layouts (CDNA5 ISA 7.12.2) ----
// A (16x32, MxK): lanes 0-15 hold M=lane, K={0..7,16..23}; lanes 16-31 hold
// M=lane-16, K={8..15,24..31}. half index within v16h = (K/16)*8 + (K%16)%8.
__device__ __forceinline__ int afrag_idx(int mt, int kc, int mm, int k) {
  int grp = k >> 4, kk = k & 15;
  int lane = mm + ((kk & 8) ? 16 : 0);
  int half = grp * 8 + (kk & 7);
  return ((mt * 8 + kc) * 32 + lane) * 16 + half;
}
// B (32x16, KxN): lanes 0-15 hold N=lane K={0..7,16..23}; lanes 16-31 hold K={8..15,24..31}.
__device__ __forceinline__ int bfrag_idx(int nt, int kc, int nn, int k) {
  int lane = nn + ((k & 8) ? 16 : 0);
  int half = (k >> 4) * 8 + (k & 7);
  return ((nt * 8 + kc) * 32 + lane) * 16 + half;
}

// One 16x16 f32 tile over K=256: 8 chained v_wmma_f32_16x16x32_f16.
// A/B pointers are in v16h units, fragment-packed: [kc][lane] contiguous,
// so each fragment load is 2x b128 (global or ds, by address space).
__device__ __forceinline__ v8f wmma_k256(const v16h* __restrict__ A,
                                         const v16h* __restrict__ B, int lane) {
  v8f c = {};
#pragma unroll
  for (int kc = 0; kc < 8; ++kc) {
    v16h a = A[kc * 32 + lane];
    v16h b = B[kc * 32 + lane];
    c = __builtin_amdgcn_wmma_f32_16x16x32_f16(false, a, false, b,
                                               (short)0, c, false, false);
  }
  return c;
}

// C/D layout: VGPR r <-> M = r + 8*(lane>=16); N = lane&15.
__device__ __forceinline__ void store_tile(float* __restrict__ dst, int ldc,
                                           int mt, int nt, int lane, v8f c) {
  int row = mt * 16 + ((lane >> 4) << 3);
  int col = nt * 16 + (lane & 15);
#pragma unroll
  for (int r = 0; r < 8; ++r) dst[(row + r) * ldc + col] = c[r];
}

__device__ __forceinline__ float sigf(float x) { return 1.0f / (1.0f + __expf(-x)); }
__device__ __forceinline__ float tanhfast(float x) { return 2.0f * sigf(2.0f * x) - 1.0f; }

#if HAVE_TDM
// Issue one TDM DMA: copy 8192 contiguous bytes global -> LDS.
// D# per CDNA5 ISA: count=1, type=2 ("image"),
// data_size=8B, tile 1024x1, tensor_dim0=1024, strides=1024. Groups 2/3
// zero (<=2D tensor). Tracked by TENSORcnt.
__device__ __forceinline__ void tdm_load_8k(const void* gsrc, void* lds_dst) {
  unsigned long long ga = (unsigned long long)(uintptr_t)gsrc;
  unsigned lo = (unsigned)(uintptr_t)lds_dst;  // low 32 bits = LDS byte offset
  v4u32 g0;
  g0[0] = 1u;                                   // count=1, user load descriptor
  g0[1] = lo;                                   // lds_addr
  g0[2] = (unsigned)ga;                         // global_addr[31:0]
  g0[3] = (unsigned)((ga >> 32) & 0x1FFFFFFull) // global_addr[56:32]
        | (2u << 30);                           // type=2
  v8i32 g1;
  g1[0] = (int)(3u << 16);                      // data_size=8B; no mcast/pad
  g1[1] = (int)(1024u << 16);                   // tensor_dim0 = 1024 (lo16)
  g1[2] = (int)(1u << 16);                      // tensor_dim0 hi; tensor_dim1=1
  g1[3] = (int)(1024u << 16);                   // tensor_dim1 hi; tile_dim0=1024
  g1[4] = 1;                                    // tile_dim1=1; tile_dim2=0
  g1[5] = 1024;                                 // tensor_dim0_stride lo32
  g1[6] = (int)(1024u << 16);                   // stride0 hi; stride1 lo16
  g1[7] = 0;                                    // stride1 hi
  v4i32 z = {};
#if defined(__clang_major__) && (__clang_major__ >= 23)
  v8i32 z8 = {};
  __builtin_amdgcn_tensor_load_to_lds(g0, g1, z, z, z8, 0);
#else
  __builtin_amdgcn_tensor_load_to_lds(g0, g1, z, z, 0);
#endif
}
#endif

// ---------------- init: zero h-state + barrier words ----------------
__global__ void k_init(unsigned* hz, int n, unsigned* sync) {
  for (int i = blockIdx.x * blockDim.x + threadIdx.x; i < n;
       i += gridDim.x * blockDim.x)
    hz[i] = 0u;
  if (blockIdx.x == 0 && threadIdx.x == 0) { sync[0] = 0u; sync[1] = 0u; }
}

// ---------------- pack x (f32 row-major) -> f16 A-fragments ----------------
__global__ void k_pack_x(const float* __restrict__ x, _Float16* __restrict__ xpk) {
  int i = blockIdx.x * blockDim.x + threadIdx.x;
  if (i >= MTOT * IN_) return;
  int row = i >> 8, col = i & 255;
  xpk[afrag_idx(row >> 4, col >> 5, row & 15, col & 31)] = (_Float16)x[i];
}

// pack W (nrows x 256 f32, row-major; out = in @ W^T) -> f16 B-fragments
__global__ void k_pack_w(const float* __restrict__ W, _Float16* __restrict__ wpk,
                         int nrows) {
  int i = blockIdx.x * blockDim.x + threadIdx.x;
  if (i >= nrows * H_) return;
  int n = i >> 8, k = i & 255;
  wpk[bfrag_idx(n >> 4, k >> 5, n & 15, k & 31)] = (_Float16)W[i];
}

// ---------------- xw0 = x @ Wx0^T (131072 x 768), 1 tile / wave -----------
__global__ void k_gemm_xw0(const v16h* __restrict__ xpk,
                           const v16h* __restrict__ wpk,
                           float* __restrict__ xw0) {
  int wid = threadIdx.x >> 5, lane = threadIdx.x & 31;
  int tile = blockIdx.x * 8 + wid;     // 8192*48 tiles
  int mt = tile / 48, nt = tile % 48;
  v8f c = wmma_k256(xpk + (size_t)mt * 256, wpk + nt * 256, lane);
  store_tile(xw0, G3, mt, nt, lane, c);
}

// ---------------- device-wide sense-reversal barrier ----------------
__device__ __forceinline__ void grid_barrier(unsigned* sync) {
  __threadfence();
  __syncthreads();
  if (threadIdx.x == 0) {
    unsigned g = __hip_atomic_load(&sync[1], __ATOMIC_RELAXED, __HIP_MEMORY_SCOPE_AGENT);
    unsigned a = __hip_atomic_fetch_add(&sync[0], 1u, __ATOMIC_ACQ_REL, __HIP_MEMORY_SCOPE_AGENT);
    if (a == RNB - 1) {
      __hip_atomic_store(&sync[0], 0u, __ATOMIC_RELAXED, __HIP_MEMORY_SCOPE_AGENT);
      __hip_atomic_fetch_add(&sync[1], 1u, __ATOMIC_RELEASE, __HIP_MEMORY_SCOPE_AGENT);
    } else {
      while (__hip_atomic_load(&sync[1], __ATOMIC_ACQUIRE, __HIP_MEMORY_SCOPE_AGENT) == g)
        __builtin_amdgcn_s_sleep(1);
    }
  }
  __syncthreads();
}

// ---------------- persistent recurrent kernel ----------------
__global__ void __launch_bounds__(RNT)
k_recurrent(const float* __restrict__ xw0,
            const _Float16* __restrict__ wh0pk, const _Float16* __restrict__ wx1pk,
            const _Float16* __restrict__ wh1pk,
            float* __restrict__ h0, float* __restrict__ h1,
            _Float16* __restrict__ h0pk, _Float16* __restrict__ h1pk,
            float* __restrict__ gh0, float* __restrict__ gx1, float* __restrict__ gh1,
            const float* __restrict__ bx0, const float* __restrict__ bh0,
            const float* __restrict__ bx1, const float* __restrict__ bh1,
            unsigned* sync) {
  const int wid = threadIdx.x >> 5, lane = threadIdx.x & 31;
  const int mt = wid;               // wave -> m-tile
  const int ntb = blockIdx.x;       // block -> n-tile column (48 of them)
  const int gtid = blockIdx.x * RNT + threadIdx.x;

  // Stage this block's loop-invariant weight slices (3 x 8KB) in LDS once.
  // B-fragments for n-tile `ntb` are contiguous: wpk + ntb*4096 halfs.
  __shared__ _Float16 sW[3 * 4096];
#if HAVE_TDM
  if (wid == 0) {                   // one wave issues the DMAs (wave-uniform)
    tdm_load_8k(wh0pk + (size_t)ntb * 4096, &sW[0]);
    tdm_load_8k(wx1pk + (size_t)ntb * 4096, &sW[4096]);
    tdm_load_8k(wh1pk + (size_t)ntb * 4096, &sW[8192]);
    __builtin_amdgcn_s_wait_tensorcnt(0);
  }
#else
  {
    const unsigned* s0 = (const unsigned*)(wh0pk + (size_t)ntb * 4096);
    const unsigned* s1 = (const unsigned*)(wx1pk + (size_t)ntb * 4096);
    const unsigned* s2 = (const unsigned*)(wh1pk + (size_t)ntb * 4096);
    unsigned* d = (unsigned*)sW;
    for (int i = threadIdx.x; i < 2048; i += RNT) {
      d[i] = s0[i]; d[2048 + i] = s1[i]; d[4096 + i] = s2[i];
    }
  }
#endif
  __syncthreads();
  const v16h* sWh0 = (const v16h*)&sW[0];
  const v16h* sWx1 = (const v16h*)&sW[4096];
  const v16h* sWh1 = (const v16h*)&sW[8192];

  for (int t = 0; t < TT; ++t) {
    // Phase A: gh0 = h0 @ Wh0^T   (A from L2, B from LDS)
    {
      v8f c = wmma_k256((const v16h*)h0pk + mt * 256, sWh0, lane);
      store_tile(gh0, G3, mt, ntb, lane, c);
    }
    grid_barrier(sync);
    // Phase B: layer-0 gates -> new h0 (f32 carry + f16 A-fragment copy)
    for (int i = gtid; i < BB * H_; i += RTHREADS) {
      int m = i >> 8, cc = i & 255;
      const float* xw = xw0 + ((size_t)(m * TT + t)) * G3;  // x row = b*T + t
      const float* gh = gh0 + m * G3;
      float r = sigf(xw[cc] + bx0[cc] + gh[cc] + bh0[cc]);
      float z = sigf(xw[cc + 256] + bx0[cc + 256] + gh[cc + 256] + bh0[cc + 256]);
      float n = tanhfast(xw[cc + 512] + bx0[cc + 512] +
                         r * (gh[cc + 512] + bh0[cc + 512]));
      float hnew = z * h0[i] + (1.0f - z) * n;
      h0[i] = hnew;
      h0pk[afrag_idx(m >> 4, cc >> 5, m & 15, cc & 31)] = (_Float16)hnew;
    }
    grid_barrier(sync);
    // Phase C: gx1 = h0 @ Wx1^T ; gh1 = h1 @ Wh1^T (2 tiles per wave)
    {
      v8f c = wmma_k256((const v16h*)h0pk + mt * 256, sWx1, lane);
      store_tile(gx1, G3, mt, ntb, lane, c);
      v8f d = wmma_k256((const v16h*)h1pk + mt * 256, sWh1, lane);
      store_tile(gh1, G3, mt, ntb, lane, d);
    }
    grid_barrier(sync);
    // Phase D: layer-1 gates -> new h1
    for (int i = gtid; i < BB * H_; i += RTHREADS) {
      int m = i >> 8, cc = i & 255;
      const float* gx = gx1 + m * G3;
      const float* gh = gh1 + m * G3;
      float r = sigf(gx[cc] + bx1[cc] + gh[cc] + bh1[cc]);
      float z = sigf(gx[cc + 256] + bx1[cc + 256] + gh[cc + 256] + bh1[cc + 256]);
      float n = tanhfast(gx[cc + 512] + bx1[cc + 512] +
                         r * (gh[cc + 512] + bh1[cc + 512]));
      float hnew = z * h1[i] + (1.0f - z) * n;
      h1[i] = hnew;
      h1pk[afrag_idx(m >> 4, cc >> 5, m & 15, cc & 31)] = (_Float16)hnew;
    }
    grid_barrier(sync);
  }
}

// ---------------- out = h1 @ Wo^T + bo : 64 tiles ----------------
__global__ void k_final(const _Float16* __restrict__ h1pk,
                        const _Float16* __restrict__ wopk,
                        const float* __restrict__ bo, float* __restrict__ out) {
  int wid = threadIdx.x >> 5, lane = threadIdx.x & 31;
  int tile = blockIdx.x * 8 + wid;    // 8 blocks * 8 waves = 64 tiles
  int mt = tile >> 3, nt = tile & 7;
  v8f c = wmma_k256((const v16h*)h1pk + mt * 256, (const v16h*)wopk + nt * 256, lane);
  int row = mt * 16 + ((lane >> 4) << 3);
  int col = nt * 16 + (lane & 15);
  float b = bo[col];
#pragma unroll
  for (int r = 0; r < 8; ++r) out[(row + r) * OUTN + col] = c[r] + b;
}

// ---------------- host launcher ----------------
extern "C" void kernel_launch(void* const* d_in, const int* in_sizes, int n_in,
                              void* d_out, int out_size, void* d_ws, size_t ws_size,
                              hipStream_t stream) {
  (void)in_sizes; (void)n_in; (void)out_size; (void)ws_size;
  const float* x   = (const float*)d_in[0];
  const float* Wx0 = (const float*)d_in[1];
  const float* bx0 = (const float*)d_in[2];
  const float* Wh0 = (const float*)d_in[3];
  const float* bh0 = (const float*)d_in[4];
  const float* Wx1 = (const float*)d_in[5];
  const float* bx1 = (const float*)d_in[6];
  const float* Wh1 = (const float*)d_in[7];
  const float* bh1 = (const float*)d_in[8];
  const float* Wo  = (const float*)d_in[9];
  const float* bo  = (const float*)d_in[10];

  char* ws = (char*)d_ws;
  constexpr size_t SZ_XPK = (size_t)MTOT * IN_ * 2;   // 67 MB
  constexpr size_t SZ_W   = (size_t)G3 * H_ * 2;      // 384 KB each
  constexpr size_t OFF_XPK = 0;
  constexpr size_t OFF_WX0 = OFF_XPK + SZ_XPK;
  constexpr size_t OFF_WH0 = OFF_WX0 + SZ_W;
  constexpr size_t OFF_WX1 = OFF_WH0 + SZ_W;
  constexpr size_t OFF_WH1 = OFF_WX1 + SZ_W;
  constexpr size_t OFF_WO  = OFF_WH1 + SZ_W;
  constexpr size_t OFF_XW0 = OFF_WO + (size_t)OUTN * H_ * 2;
  constexpr size_t OFF_H0  = OFF_XW0 + (size_t)MTOT * G3 * 4;  // 402 MB xw0
  constexpr size_t OFF_H1  = OFF_H0 + (size_t)BB * H_ * 4;
  constexpr size_t OFF_H0P = OFF_H1 + (size_t)BB * H_ * 4;
  constexpr size_t OFF_H1P = OFF_H0P + (size_t)BB * H_ * 2;
  constexpr size_t OFF_GH0 = OFF_H1P + (size_t)BB * H_ * 2;
  constexpr size_t OFF_GX1 = OFF_GH0 + (size_t)BB * G3 * 4;
  constexpr size_t OFF_GH1 = OFF_GX1 + (size_t)BB * G3 * 4;
  constexpr size_t OFF_SYNC = OFF_GH1 + (size_t)BB * G3 * 4;

  _Float16* xpk   = (_Float16*)(ws + OFF_XPK);
  _Float16* wx0pk = (_Float16*)(ws + OFF_WX0);
  _Float16* wh0pk = (_Float16*)(ws + OFF_WH0);
  _Float16* wx1pk = (_Float16*)(ws + OFF_WX1);
  _Float16* wh1pk = (_Float16*)(ws + OFF_WH1);
  _Float16* wopk  = (_Float16*)(ws + OFF_WO);
  float*    xw0   = (float*)(ws + OFF_XW0);
  float*    h0    = (float*)(ws + OFF_H0);
  float*    h1    = (float*)(ws + OFF_H1);
  _Float16* h0pk  = (_Float16*)(ws + OFF_H0P);
  _Float16* h1pk  = (_Float16*)(ws + OFF_H1P);
  float*    gh0   = (float*)(ws + OFF_GH0);
  float*    gx1   = (float*)(ws + OFF_GX1);
  float*    gh1   = (float*)(ws + OFF_GH1);
  unsigned* sync  = (unsigned*)(ws + OFF_SYNC);

  // 1) zero recurrent state + barrier words (ws is poisoned, not re-poisoned)
  k_init<<<64, 256, 0, stream>>>((unsigned*)(ws + OFF_H0),
                                 (int)((OFF_GH0 - OFF_H0) / 4), sync);
  // 2) fragment-pack x and all weights to f16 once
  k_pack_x<<<(MTOT * IN_) / 256, 256, 0, stream>>>(x, xpk);
  k_pack_w<<<(G3 * H_) / 256, 256, 0, stream>>>(Wx0, wx0pk, G3);
  k_pack_w<<<(G3 * H_) / 256, 256, 0, stream>>>(Wh0, wh0pk, G3);
  k_pack_w<<<(G3 * H_) / 256, 256, 0, stream>>>(Wx1, wx1pk, G3);
  k_pack_w<<<(G3 * H_) / 256, 256, 0, stream>>>(Wh1, wh1pk, G3);
  k_pack_w<<<(OUTN * H_) / 256, 256, 0, stream>>>(Wo, wopk, OUTN);
  // 3) hoisted time-independent GEMM: xw0 = x @ Wx0^T  (393216 tiles)
  k_gemm_xw0<<<(8192 * 48) / 8, 256, 0, stream>>>((const v16h*)xpk,
                                                  (const v16h*)wx0pk, xw0);
  // 4) persistent 1024-step GRU loop (device-wide barriers between phases,
  //    TDM-staged weights in LDS)
  k_recurrent<<<RNB, RNT, 0, stream>>>(xw0, wh0pk, wx1pk, wh1pk, h0, h1,
                                       h0pk, h1pk, gh0, gx1, gh1,
                                       bx0, bh0, bx1, bh1, sync);
  // 5) output projection
  k_final<<<8, 256, 0, stream>>>(h1pk, wopk, bo, (float*)d_out);
}
